// GAT_actor_55327768708415
// MI455X (gfx1250) — matrix operand order repaired and checked by reading, not compiled, not run on hardware
//
#include <hip/hip_runtime.h>

// ---------------------------------------------------------------------------
// Types / helpers
// ---------------------------------------------------------------------------
typedef __bf16 bf16_t;
typedef bf16_t v16bf __attribute__((ext_vector_type(16)));
typedef bf16_t v8bf  __attribute__((ext_vector_type(8)));
typedef float  v8f   __attribute__((ext_vector_type(8)));
typedef unsigned short u16x8 __attribute__((ext_vector_type(8)));
typedef unsigned short ubf16;   // bf16 raw-bits storage

__device__ __forceinline__ ubf16 f2bf(float f) {
    unsigned u = __builtin_bit_cast(unsigned, f);
    unsigned r = u + 0x7FFFu + ((u >> 16) & 1u);      // round-to-nearest-even
    return (ubf16)(r >> 16);
}
__device__ __forceinline__ float bf2f(ubf16 h) {
    return __builtin_bit_cast(float, (unsigned)h << 16);
}

// Async DMA: copy 16 bytes global -> LDS via GLOBAL_LOAD_ASYNC_TO_LDS_B128
// (ASYNCcnt-tracked, GVS addressing: saddr base + 32-bit voffset).
// lds_byte_off is relative to the dynamic-LDS base (no static LDS in these
// kernels, so it equals the wave's LDS byte address).
__device__ __forceinline__ void async_g2l_b128(unsigned lds_byte_off,
                                               const void* gbase,
                                               unsigned gbyte_off) {
    asm volatile("global_load_async_to_lds_b128 %0, %1, %2"
                 :
                 : "v"(lds_byte_off), "v"(gbyte_off), "s"(gbase)
                 : "memory");
}
__device__ __forceinline__ void async_wait0() {
    asm volatile("s_wait_asynccnt 0" ::: "memory");
}

// A fragment (16x32 bf16, M=rows of X): lane l holds row (l&15),
// elems 0..7 = K kb..kb+7, elems 8..15 = K kb+16..kb+23, kb=(l>>4)*8.
__device__ __forceinline__ v16bf lds_load_a(const ubf16* base, int row0, int k0,
                                            int ld, int lane) {
    const ubf16* p = base + (size_t)(row0 + (lane & 15)) * ld
                          + (k0 + ((lane >> 4) << 3));
    u16x8 lo = *(const u16x8*)p;
    u16x8 hi = *(const u16x8*)(p + 16);
    v8bf l = __builtin_bit_cast(v8bf, lo);
    v8bf h = __builtin_bit_cast(v8bf, hi);
    return __builtin_shufflevector(l, h, 0,1,2,3,4,5,6,7,8,9,10,11,12,13,14,15);
}
// B fragment (32x16 bf16) from transposed weights Wt[n][k]: lane l holds
// column (l&15), elems 0..7 = K kb..kb+7, elems 8..15 = K kb+16..kb+23, kb=(l>>4)*8.
__device__ __forceinline__ v16bf lds_load_b(const ubf16* baseT, int col0, int k0,
                                            int ldk, int lane) {
    const ubf16* p = baseT + (size_t)(col0 + (lane & 15)) * ldk
                           + (k0 + ((lane >> 4) << 3));
    u16x8 lo = *(const u16x8*)p;
    u16x8 hi = *(const u16x8*)(p + 16);
    v8bf l = __builtin_bit_cast(v8bf, lo);
    v8bf h = __builtin_bit_cast(v8bf, hi);
    return __builtin_shufflevector(l, h, 0,1,2,3,4,5,6,7,8,9,10,11,12,13,14,15);
}
__device__ __forceinline__ v8f wmma_bf16(v16bf a, v16bf b, v8f c) {
    return __builtin_amdgcn_wmma_f32_16x16x32_bf16(false, a, false, b,
                                                   (short)0, c, false, false);
}

// ---------------------------------------------------------------------------
// Problem constants
// ---------------------------------------------------------------------------
#define NBATCH   65536
#define NNODE    9
#define NHID     256
#define NHEAD    8
#define CPH      32          // channels per head (all three GAT layers)
#define FC2_IN   2560        // 9*256 + 256
#define FC2_OUT  512

// workspace layout (bytes)
#define OFF_W1T   0u                     // [256][64]  bf16 =   32768
#define OFF_W2T   32768u                 // [256][256] bf16 =  131072
#define OFF_W3T   163840u                // [256][256] bf16 =  131072
#define OFF_WF1T  294912u                // [256][128] bf16 =   65536
#define OFF_WF2T  360448u                // [512][2560]bf16 = 2621440
#define OFF_BUFA  4194304u               // act1 / fc2in : 65536*2560*2 = 335544320
#define OFF_BUFB  339738624u             // act2 / fc2out: 589824*256*2 = 301989888

// ---------------------------------------------------------------------------
// Weight convert + transpose: dst[n*K + k] = bf16(src[k*N + n])
// ---------------------------------------------------------------------------
__global__ __launch_bounds__(256)
void transpose_bf16_kernel(const float* __restrict__ src, ubf16* __restrict__ dst,
                           int K, int N) {
    int total = K * N;
    for (int i = blockIdx.x * blockDim.x + threadIdx.x; i < total;
         i += gridDim.x * blockDim.x) {
        int n = i / K, k = i - n * K;
        dst[i] = f2bf(src[(size_t)k * N + n]);
    }
}

// ---------------------------------------------------------------------------
// Fused GAT layer: h = X@W + b ; attention softmax over 9 neighbors ; ELU.
// One block = 16 graphs = 144 rows. Whole Wt lives in LDS.
// ---------------------------------------------------------------------------
template<int CIN, bool IN_F32>
__global__ __launch_bounds__(256)
void gat_layer_kernel(const void* __restrict__ in_p,
                      const float* __restrict__ adj,     // [B,9,9]
                      const ubf16* __restrict__ Wt,      // [256][CIN] bf16
                      const float* __restrict__ bias,    // [256]
                      const float* __restrict__ avec,    // [8][64]
                      ubf16* __restrict__ out,           // bf16
                      int in_bstride, int out_bstride) {
    extern __shared__ char smem[];
    ubf16* sW   = (ubf16*)smem;                         // 256*CIN
    ubf16* sX   = sW + 256 * CIN;                       // 144*CIN
    ubf16* sH   = sX + 144 * CIN;                       // 144*256
    float* sB   = (float*)(sH + 144 * 256);             // 256
    float* sA   = sB + 256;                             // 512
    float* sAl  = sA + 512;                             // 1152
    float* sAr  = sAl + 1152;                           // 1152
    float* sAdj = sAr + 1152;                           // 16*81

    const int tid  = threadIdx.x;
    const int lane = tid & 31, wave = tid >> 5;
    const int b0   = blockIdx.x * 16;
    const unsigned sW_off = 0u;                         // LDS byte offsets
    const unsigned sX_off = (unsigned)(256 * CIN * 2);

    // stage weights: async DMA global -> LDS, 16B per op
    for (int i = tid; i < 256 * CIN / 8; i += 256)
        async_g2l_b128(sW_off + (unsigned)i * 16u, Wt, (unsigned)i * 16u);

    if (tid < 256) sB[tid] = bias[tid];
    for (int i = tid; i < 512; i += 256) sA[i] = avec[i];
    for (int i = tid; i < 16 * 81; i += 256) sAdj[i] = adj[(size_t)b0 * 81 + i];

    if constexpr (IN_F32) {   // layer 1: convert obs_gan f32 -> bf16 (VGPR path)
        const float* g = (const float*)in_p;
        for (int i = tid; i < 144 * CIN; i += 256) {
            int r = i / CIN, k = i - r * CIN;
            int b = b0 + r / 9, n = r - (r / 9) * 9;
            sX[i] = f2bf(g[(size_t)b * in_bstride + n * CIN + k]);
        }
    } else {                  // layers 2/3: bf16 activations, async DMA
        const int vpr = CIN / 8;
        for (int i = tid; i < 144 * vpr; i += 256) {
            int r = i / vpr, v = i - r * vpr;
            int b = b0 + r / 9, n = r - (r / 9) * 9;
            unsigned goff = ((unsigned)b * (unsigned)in_bstride
                             + (unsigned)(n * CIN + v * 8)) * 2u;
            async_g2l_b128(sX_off + (unsigned)i * 16u, in_p, goff);
        }
    }
    async_wait0();
    __syncthreads();

    // ---- GEMM: 9 row-tiles x 16 col-tiles = 144 WMMA tiles over 8 waves ----
    for (int t = wave; t < 144; t += 8) {
        int rt = t >> 4, ct = t & 15;
        v8f acc = {0.f, 0.f, 0.f, 0.f, 0.f, 0.f, 0.f, 0.f};
#pragma unroll
        for (int k0 = 0; k0 < CIN; k0 += 32)
            acc = wmma_bf16(lds_load_a(sX, rt * 16, k0, CIN, lane),
                            lds_load_b(sW, ct * 16, k0, CIN, lane), acc);
        int col = ct * 16 + (lane & 15);
        float bv = sB[col];
        int r0 = rt * 16 + ((lane >> 4) << 3);
#pragma unroll
        for (int r = 0; r < 8; ++r)
            sH[(r0 + r) * 256 + col] = f2bf(acc[r] + bv);
    }
    __syncthreads();

    // ---- al/ar = h . a  (per row, per head) ----
    for (int i = tid; i < 1152; i += 256) {
        int row = i >> 3, h = i & 7;
        const ubf16* hp = sH + row * 256 + h * CPH;
        const float* ap = sA + h * 64;
        float sl = 0.f, sr = 0.f;
#pragma unroll
        for (int c = 0; c < CPH; ++c) {
            float hv = bf2f(hp[c]);
            sl += hv * ap[c];
            sr += hv * ap[CPH + c];
        }
        sAl[i] = sl;
        sAr[i] = sr;
    }
    __syncthreads();

    // ---- attention softmax + aggregation + ELU ----
    for (int it = tid; it < 1152; it += 256) {   // 16 graphs * 9 nodes * 8 heads
        int bb = it / 72, p = it - bb * 72;
        int i = p >> 3, h = p & 7;
        int rrow0 = bb * 9;
        float ali = sAl[(rrow0 + i) * 8 + h];
        float w[9];
        unsigned msk = 0;
        float m = -3.0e38f;
#pragma unroll
        for (int j = 0; j < 9; ++j) {
            float ad = sAdj[bb * 81 + i * 9 + j];
            float t = 0.f;
            if (ad > 0.f) {
                t = ali + sAr[(rrow0 + j) * 8 + h];
                t = t > 0.f ? t : 0.2f * t;           // leaky_relu(0.2)
                msk |= 1u << j;
                if (t > m) m = t;
            }
            w[j] = t;
        }
        float s = 0.f;
#pragma unroll
        for (int j = 0; j < 9; ++j) {
            float e = ((msk >> j) & 1u) ? __expf(w[j] - m) : 0.f;
            w[j] = e;
            s += e;
        }
        float inv = 1.f / s;
#pragma unroll
        for (int j = 0; j < 9; ++j) w[j] *= inv;

        float acc[CPH];
#pragma unroll
        for (int c = 0; c < CPH; ++c) acc[c] = 0.f;
        for (int j = 0; j < 9; ++j) {
            float wj = w[j];
            const ubf16* hp = sH + (rrow0 + j) * 256 + h * CPH;
#pragma unroll
            for (int c = 0; c < CPH; ++c) acc[c] += wj * bf2f(hp[c]);
        }
        ubf16* orow = out + (size_t)(b0 + bb) * out_bstride + i * 256 + h * CPH;
#pragma unroll
        for (int c = 0; c < CPH; c += 2) {           // ELU + packed bf16 store
            float o0 = acc[c]     > 0.f ? acc[c]     : (__expf(acc[c])     - 1.f);
            float o1 = acc[c + 1] > 0.f ? acc[c + 1] : (__expf(acc[c + 1]) - 1.f);
            unsigned pk = (unsigned)f2bf(o0) | ((unsigned)f2bf(o1) << 16);
            *(unsigned*)(orow + c) = pk;
        }
    }
}

// ---------------------------------------------------------------------------
// fc1: tanh(obs_fc @ Wfc1 + b) -> bf16 into fc2in columns [2304,2560)
// ---------------------------------------------------------------------------
__global__ __launch_bounds__(256)
void fc1_kernel(const float* __restrict__ x,        // [B,128] f32
                const ubf16* __restrict__ Wt,       // [256][128] bf16
                const float* __restrict__ bias,     // [256]
                ubf16* __restrict__ out) {          // fc2in [B][2560] bf16
    extern __shared__ char smem[];
    ubf16* sW = (ubf16*)smem;                       // 256*128
    ubf16* sX = sW + 256 * 128;                     // 64*128
    float* sB = (float*)(sX + 64 * 128);            // 256
    const int tid = threadIdx.x, lane = tid & 31, wave = tid >> 5;
    const int r0g = blockIdx.x * 64;

    // async weight staging
    for (int i = tid; i < 256 * 128 / 8; i += 256)
        async_g2l_b128((unsigned)i * 16u, Wt, (unsigned)i * 16u);
    if (tid < 256) sB[tid] = bias[tid];
    for (int i = tid; i < 64 * 128; i += 256) {
        int r = i >> 7, k = i & 127;
        sX[i] = f2bf(x[(size_t)(r0g + r) * 128 + k]);
    }
    async_wait0();
    __syncthreads();

    for (int t = wave; t < 64; t += 8) {            // 4 row-tiles x 16 col-tiles
        int rt = t >> 4, ct = t & 15;
        v8f acc = {0.f, 0.f, 0.f, 0.f, 0.f, 0.f, 0.f, 0.f};
#pragma unroll
        for (int k0 = 0; k0 < 128; k0 += 32)
            acc = wmma_bf16(lds_load_a(sX, rt * 16, k0, 128, lane),
                            lds_load_b(sW, ct * 16, k0, 128, lane), acc);
        int col = ct * 16 + (lane & 15);
        float bv = sB[col];
        int rr0 = rt * 16 + ((lane >> 4) << 3);
#pragma unroll
        for (int r = 0; r < 8; ++r) {
            float v = tanhf(acc[r] + bv);
            out[(size_t)(r0g + rr0 + r) * FC2_IN + 2304 + col] = f2bf(v);
        }
    }
}

// ---------------------------------------------------------------------------
// fc2: tanh([B,2560] @ [2560,512] + b) -> bf16. K-chunked async LDS staging.
// ---------------------------------------------------------------------------
__global__ __launch_bounds__(256)
void fc2_kernel(const ubf16* __restrict__ x,        // [B][2560] bf16
                const ubf16* __restrict__ Wt,       // [512][2560] bf16
                const float* __restrict__ bias,     // [512]
                ubf16* __restrict__ out) {          // [B][512] bf16
    extern __shared__ char smem[];
    ubf16* sW = (ubf16*)smem;                       // 512*128
    ubf16* sX = sW + 512 * 128;                     // 64*128
    float* sB = (float*)(sX + 64 * 128);            // 512
    const int tid = threadIdx.x, lane = tid & 31, wave = tid >> 5;
    const int r0g = blockIdx.x * 64;
    const int rt = wave & 3;                        // 4 row-tiles
    const int cb = (wave >> 2) * 16;                // col-tile base: 0 or 16
    const ubf16* xblk = x + (size_t)r0g * FC2_IN;   // block-uniform base
    const unsigned sX_off = 512u * 128u * 2u;

    for (int i = tid; i < 512; i += 256) sB[i] = bias[i];

    v8f acc[16];
#pragma unroll
    for (int i = 0; i < 16; ++i) acc[i] = (v8f){0.f,0.f,0.f,0.f,0.f,0.f,0.f,0.f};

    for (int kc = 0; kc < FC2_IN; kc += 128) {
        __syncthreads();                            // prior chunk's compute done
        // stage 512x128 weight chunk (async DMA)
        for (int i = tid; i < 512 * 16; i += 256) {
            int n = i >> 4, v = i & 15;
            unsigned goff = ((unsigned)n * FC2_IN + (unsigned)(kc + v * 8)) * 2u;
            async_g2l_b128((unsigned)i * 16u, Wt, goff);
        }
        // stage 64x128 activation chunk (async DMA)
        for (int i = tid; i < 64 * 16; i += 256) {
            int r = i >> 4, v = i & 15;
            unsigned goff = ((unsigned)r * FC2_IN + (unsigned)(kc + v * 8)) * 2u;
            async_g2l_b128(sX_off + (unsigned)i * 16u, xblk, goff);
        }
        if (kc + 128 < FC2_IN)                      // warm L2 for next chunk
            __builtin_prefetch(Wt + (size_t)(tid * 2) * FC2_IN + kc + 128, 0, 1);
        async_wait0();
        __syncthreads();
#pragma unroll
        for (int ks = 0; ks < 128; ks += 32) {
            v16bf a = lds_load_a(sX, rt * 16, ks, 128, lane);   // A reused x16
#pragma unroll
            for (int c = 0; c < 16; ++c) {
                v16bf b = lds_load_b(sW, (cb + c) * 16, ks, 128, lane);
                acc[c] = wmma_bf16(a, b, acc[c]);
            }
        }
    }

    int rr0 = r0g + rt * 16 + ((lane >> 4) << 3);
#pragma unroll
    for (int c = 0; c < 16; ++c) {
        int col = (cb + c) * 16 + (lane & 15);
        float bv = sB[col];
#pragma unroll
        for (int r = 0; r < 8; ++r) {
            float v = tanhf(acc[c][r] + bv);
            out[(size_t)(rr0 + r) * FC2_OUT + col] = f2bf(v);
        }
    }
}

// ---------------------------------------------------------------------------
// fc3: [B,512] @ [512,21] + b, softmax over groups of 3 -> f32 out
// ---------------------------------------------------------------------------
__global__ __launch_bounds__(256)
void fc3_kernel(const ubf16* __restrict__ x,        // [B][512] bf16
                const float* __restrict__ W,        // [512,21] f32
                const float* __restrict__ bias,     // [21]
                float* __restrict__ out) {          // [B][21] f32
    extern __shared__ char smem[];
    float* sW = (float*)smem;                       // 512*21
    const int tid = threadIdx.x;
    for (int i = tid; i < 512 * 21; i += 256) sW[i] = W[i];
    __syncthreads();

    const int row = blockIdx.x * 256 + tid;
    float acc[21];
#pragma unroll
    for (int j = 0; j < 21; ++j) acc[j] = bias[j];
    const ubf16* xr = x + (size_t)row * 512;
    for (int k = 0; k < 512; ++k) {
        float xv = bf2f(xr[k]);
        const float* wr = sW + k * 21;
#pragma unroll
        for (int j = 0; j < 21; ++j) acc[j] += xv * wr[j];
    }
    float* orow = out + (size_t)row * 21;
#pragma unroll
    for (int g = 0; g < 7; ++g) {
        float a0 = acc[g * 3], a1 = acc[g * 3 + 1], a2 = acc[g * 3 + 2];
        float m = fmaxf(a0, fmaxf(a1, a2));
        float e0 = __expf(a0 - m), e1 = __expf(a1 - m), e2 = __expf(a2 - m);
        float inv = 1.f / (e0 + e1 + e2);
        orow[g * 3] = e0 * inv;
        orow[g * 3 + 1] = e1 * inv;
        orow[g * 3 + 2] = e2 * inv;
    }
}

// ---------------------------------------------------------------------------
// Host launcher
// ---------------------------------------------------------------------------
extern "C" void kernel_launch(void* const* d_in, const int* in_sizes, int n_in,
                              void* d_out, int out_size, void* d_ws, size_t ws_size,
                              hipStream_t stream) {
    (void)in_sizes; (void)n_in; (void)out_size; (void)ws_size;
    const float* obs_gan = (const float*)d_in[0];
    const float* obs_fc  = (const float*)d_in[1];
    const float* adj     = (const float*)d_in[2];
    const float* W1   = (const float*)d_in[3];
    const float* b1   = (const float*)d_in[4];
    const float* a1   = (const float*)d_in[5];
    const float* W2   = (const float*)d_in[6];
    const float* b2   = (const float*)d_in[7];
    const float* a2   = (const float*)d_in[8];
    const float* W3   = (const float*)d_in[9];
    const float* b3   = (const float*)d_in[10];
    const float* a3   = (const float*)d_in[11];
    const float* Wfc1 = (const float*)d_in[12];
    const float* bfc1 = (const float*)d_in[13];
    const float* Wfc2 = (const float*)d_in[14];
    const float* bfc2 = (const float*)d_in[15];
    const float* Wfc3 = (const float*)d_in[16];
    const float* bfc3 = (const float*)d_in[17];

    char* ws = (char*)d_ws;
    ubf16* W1t   = (ubf16*)(ws + OFF_W1T);
    ubf16* W2t   = (ubf16*)(ws + OFF_W2T);
    ubf16* W3t   = (ubf16*)(ws + OFF_W3T);
    ubf16* Wfc1t = (ubf16*)(ws + OFF_WF1T);
    ubf16* Wfc2t = (ubf16*)(ws + OFF_WF2T);
    ubf16* bufA  = (ubf16*)(ws + OFF_BUFA);   // act1, then fc2in [B][2560]
    ubf16* bufB  = (ubf16*)(ws + OFF_BUFB);   // act2, then fc2out [B][512]

    // weight convert+transpose (bf16, [N][K])
    transpose_bf16_kernel<<<64,   256, 0, stream>>>(W1,   W1t,   64,   256);
    transpose_bf16_kernel<<<256,  256, 0, stream>>>(W2,   W2t,   256,  256);
    transpose_bf16_kernel<<<256,  256, 0, stream>>>(W3,   W3t,   256,  256);
    transpose_bf16_kernel<<<128,  256, 0, stream>>>(Wfc1, Wfc1t, 128,  256);
    transpose_bf16_kernel<<<4096, 256, 0, stream>>>(Wfc2, Wfc2t, 2560, 512);

    // dynamic LDS sizes
    const int smem1  = 142400;   // gat CIN=64
    const int smem23 = 296000;   // gat CIN=256 (fits 320KB WGP LDS)
    const int smemf1 = 82944;
    const int smemf2 = 149504;
    const int smemf3 = 43008;
    (void)hipFuncSetAttribute((const void*)&gat_layer_kernel<64, true>,
        hipFuncAttributeMaxDynamicSharedMemorySize, smem1);
    (void)hipFuncSetAttribute((const void*)&gat_layer_kernel<256, false>,
        hipFuncAttributeMaxDynamicSharedMemorySize, smem23);
    (void)hipFuncSetAttribute((const void*)&fc2_kernel,
        hipFuncAttributeMaxDynamicSharedMemorySize, smemf2);
    (void)hipFuncSetAttribute((const void*)&fc1_kernel,
        hipFuncAttributeMaxDynamicSharedMemorySize, smemf1);

    // GAT stack: L1 obs->bufA, L2 bufA->bufB, L3 bufB->bufA(as fc2in, stride 2560)
    gat_layer_kernel<64, true><<<4096, 256, smem1, stream>>>(
        (const void*)obs_gan, adj, W1t, b1, a1, bufA, 576, 2304);
    gat_layer_kernel<256, false><<<4096, 256, smem23, stream>>>(
        (const void*)bufA, adj, W2t, b2, a2, bufB, 2304, 2304);
    gat_layer_kernel<256, false><<<4096, 256, smem23, stream>>>(
        (const void*)bufB, adj, W3t, b3, a3, bufA, 2304, 2560);

    // MLP head
    fc1_kernel<<<1024, 256, smemf1, stream>>>(obs_fc, Wfc1t, bfc1, bufA);
    fc2_kernel<<<1024, 256, smemf2, stream>>>(bufA, Wfc2t, bfc2, bufB);
    fc3_kernel<<<256,  256, smemf3, stream>>>(bufB, Wfc3, bfc3, (float*)d_out);
}
